// CosineDistanceAttention_7189775253877
// MI455X (gfx1250) — matrix operand order repaired and checked by reading, not compiled, not run on hardware
//
#include <hip/hip_runtime.h>
#include <hip/hip_bf16.h>

// ---- problem constants ----
#define B_ 16
#define L_ 4096
#define N_ 512
#define D_ 256
#define EPS_ 1e-8f
#define NEG_MAX_ (-3.402823466e38f)

typedef __attribute__((ext_vector_type(16))) __bf16 v16bf;
typedef __attribute__((ext_vector_type(8)))  __bf16 v8bf;
typedef __attribute__((ext_vector_type(8)))  float  v8f;

__device__ __forceinline__ __bf16 f2bf(float f) { return (__bf16)f; }
__device__ __forceinline__ float  fastrcp(float x) { return __builtin_amdgcn_rcpf(x); }

// A-fragment (16-bit 16x32, M=lane%16): element e -> K = hi*16 + e.
// Two contiguous 16B chunks.
__device__ __forceinline__ v16bf afrag(const __bf16* rowp, int kb, int hi) {
  v8bf lo = *(const v8bf*)(rowp + kb + hi * 16);
  v8bf hv = *(const v8bf*)(rowp + kb + hi * 16 + 8);
  return __builtin_shufflevector(lo, hv, 0, 1, 2, 3, 4, 5, 6, 7,
                                 8, 9, 10, 11, 12, 13, 14, 15);
}

// ---------------------------------------------------------------------------
// Prep: per-profile-row L2 norms + P converted to bf16 in two layouts:
//   Prow (B,N,D)  -- B-fragments of GEMM1 (sims)
//   Pt   (B,D,N)  -- B-fragments of GEMM2 (embedding)
// ---------------------------------------------------------------------------
__global__ __launch_bounds__(256) void prep_kernel(const float* __restrict__ P,
                                                   float* __restrict__ pn,
                                                   __bf16* __restrict__ Prow,
                                                   __bf16* __restrict__ Pt) {
  const int tid = threadIdx.x;
  const int row = blockIdx.x * 32 + (tid >> 3);   // global profile row, 8 thr/row
  const int sub = tid & 7;
  const int b = row >> 9;                          // row / N_
  const int n = row & (N_ - 1);
  const float* p  = P    + (size_t)row * D_ + sub * 32;
  __bf16*     pr  = Prow + (size_t)row * D_ + sub * 32;
  float s = 0.f;
#pragma unroll
  for (int i = 0; i < 32; ++i) {
    float v = p[i];
    s += v * v;
    __bf16 bv = f2bf(v);
    pr[i] = bv;
    Pt[((size_t)b * D_ + sub * 32 + i) * N_ + n] = bv;
  }
  s += __shfl_xor(s, 1, 32);
  s += __shfl_xor(s, 2, 32);
  s += __shfl_xor(s, 4, 32);
  if (sub == 0) pn[row] = sqrtf(s);
}

// ---------------------------------------------------------------------------
// Fused: sims GEMM (bf16 WMMA) + cosine scale + masked softmax -> weights
//        + embedding GEMM (bf16 WMMA), one block per (b, 32-row L tile).
// 8 waves: GEMM1 each wave owns a 64-wide N strip; GEMM2 a 32-wide D strip.
// ---------------------------------------------------------------------------
__global__ __launch_bounds__(256) void fused_kernel(
    const float*  __restrict__ X,      // (B,L,D) f32
    const __bf16* __restrict__ Prow,   // (B,N,D) bf16
    const __bf16* __restrict__ Pt,     // (B,D,N) bf16
    const int*    __restrict__ lens,   // (B,)
    const float*  __restrict__ pnorm,  // (B,N)
    float* __restrict__ Wout,          // (B,L,N)
    float* __restrict__ Eout)          // (B,L,D)
{
  __shared__ __bf16 Xs[32][D_];    // 16 KB : X tile, bf16
  __shared__ __bf16 Wsh[32][N_];   // 32 KB : weights tile, bf16 (GEMM2 A)
  __shared__ float  xnorm[32];
  __shared__ float  red[8][32];    // cross-wave row reductions

  const int tid = threadIdx.x;
  const int b  = blockIdx.x >> 7;            // L_/32 = 128 tiles per batch
  const int lt = (blockIdx.x & 127) * 32;

  // ---- stage X tile (f32 -> bf16) + row norms ----
  {
    const int r   = tid >> 3;
    const int sub = tid & 7;
    const float* xp = X + ((size_t)b * L_ + lt + r) * D_ + sub * 32;
    float s = 0.f;
#pragma unroll
    for (int i = 0; i < 32; ++i) {
      float v = xp[i];
      s += v * v;
      Xs[r][sub * 32 + i] = f2bf(v);
    }
    s += __shfl_xor(s, 1, 32);
    s += __shfl_xor(s, 2, 32);
    s += __shfl_xor(s, 4, 32);
    if (sub == 0) xnorm[r] = sqrtf(s);
  }
  __syncthreads();

  const int wave = tid >> 5;
  const int lane = tid & 31;
  const int hi   = lane >> 4;
  const int ln   = lane & 15;
  const int nbase = wave * 64;

  // ================= GEMM1: S(32x512) = Xtile @ P^T =================
  v8f acc[2][4];
#pragma unroll
  for (int mt = 0; mt < 2; ++mt)
#pragma unroll
    for (int nt = 0; nt < 4; ++nt)
#pragma unroll
      for (int i = 0; i < 8; ++i) acc[mt][nt][i] = 0.f;

  const __bf16* prow_b = Prow + (size_t)b * N_ * D_;
#pragma unroll 1
  for (int kk = 0; kk < 8; ++kk) {
    const int kb = kk * 32;
    v16bf a[2];
    a[0] = afrag(&Xs[ln][0], kb, hi);
    a[1] = afrag(&Xs[16 + ln][0], kb, hi);
#pragma unroll
    for (int nt = 0; nt < 4; ++nt) {
      const int n = nbase + nt * 16 + ln;
      // B: lane=N, element e -> K = kb + hi*16 + e  (16 contiguous bf16)
      v16bf bfv = *(const v16bf*)(prow_b + (size_t)n * D_ + kb + hi * 16);
#pragma unroll
      for (int mt = 0; mt < 2; ++mt)
        acc[mt][nt] = __builtin_amdgcn_wmma_f32_16x16x32_bf16(
            false, a[mt], false, bfv, (short)0, acc[mt][nt], false, false);
    }
  }

  // ---- cosine scale (fast rcp) + pad mask ----
  const int len = lens[b];
  float pnv[4]; int valid[4];
#pragma unroll
  for (int nt = 0; nt < 4; ++nt) {
    int n = nbase + nt * 16 + ln;
    pnv[nt]   = pnorm[b * N_ + n];
    valid[nt] = (n < len);
  }
#pragma unroll
  for (int mt = 0; mt < 2; ++mt)
#pragma unroll
    for (int r = 0; r < 8; ++r) {
      float xn = xnorm[mt * 16 + hi * 8 + r];
#pragma unroll
      for (int nt = 0; nt < 4; ++nt) {
        float s = acc[mt][nt][r] * fastrcp(fmaxf(xn * pnv[nt], EPS_));
        acc[mt][nt][r] = valid[nt] ? s : NEG_MAX_;
      }
    }

  // ---- row max: in-reg -> shfl width16 -> cross-wave via LDS ----
  float mx[2][8];
#pragma unroll
  for (int mt = 0; mt < 2; ++mt)
#pragma unroll
    for (int r = 0; r < 8; ++r) {
      float m = fmaxf(fmaxf(acc[mt][0][r], acc[mt][1][r]),
                      fmaxf(acc[mt][2][r], acc[mt][3][r]));
      m = fmaxf(m, __shfl_xor(m, 1, 32));
      m = fmaxf(m, __shfl_xor(m, 2, 32));
      m = fmaxf(m, __shfl_xor(m, 4, 32));
      m = fmaxf(m, __shfl_xor(m, 8, 32));
      mx[mt][r] = m;
    }
  if (ln == 0)
#pragma unroll
    for (int mt = 0; mt < 2; ++mt)
#pragma unroll
      for (int r = 0; r < 8; ++r)
        red[wave][mt * 16 + hi * 8 + r] = mx[mt][r];
  __syncthreads();

  float rmax[2][8];
#pragma unroll
  for (int mt = 0; mt < 2; ++mt)
#pragma unroll
    for (int r = 0; r < 8; ++r) {
      int ri = mt * 16 + hi * 8 + r;
      float m = red[0][ri];
#pragma unroll
      for (int w = 1; w < 8; ++w) m = fmaxf(m, red[w][ri]);
      rmax[mt][r] = m;
    }

  // ---- exp + row sum ----
  float sm[2][8];
#pragma unroll
  for (int mt = 0; mt < 2; ++mt)
#pragma unroll
    for (int r = 0; r < 8; ++r) {
      float s = 0.f;
#pragma unroll
      for (int nt = 0; nt < 4; ++nt) {
        float e = __expf(acc[mt][nt][r] - rmax[mt][r]);  // masked -> 0
        acc[mt][nt][r] = e;
        s += e;
      }
      s += __shfl_xor(s, 1, 32);
      s += __shfl_xor(s, 2, 32);
      s += __shfl_xor(s, 4, 32);
      s += __shfl_xor(s, 8, 32);
      sm[mt][r] = s;
    }
  __syncthreads();   // all waves done reading maxes from red
  if (ln == 0)
#pragma unroll
    for (int mt = 0; mt < 2; ++mt)
#pragma unroll
      for (int r = 0; r < 8; ++r)
        red[wave][mt * 16 + hi * 8 + r] = sm[mt][r];
  __syncthreads();

  // ---- normalize; store f32 weights to global, bf16 weights to LDS ----
#pragma unroll
  for (int mt = 0; mt < 2; ++mt)
#pragma unroll
    for (int r = 0; r < 8; ++r) {
      int ri = mt * 16 + hi * 8 + r;
      float s = 0.f;
#pragma unroll
      for (int w = 0; w < 8; ++w) s += red[w][ri];
      float inv = fastrcp(s);
      size_t rowoff = ((size_t)b * L_ + lt + ri) * N_;
#pragma unroll
      for (int nt = 0; nt < 4; ++nt) {
        float wv = acc[mt][nt][r] * inv;
        Wout[rowoff + nbase + nt * 16 + ln] = wv;
        Wsh[ri][nbase + nt * 16 + ln] = f2bf(wv);
      }
    }
  __syncthreads();   // Wsh complete before GEMM2

  // ================= GEMM2: E(32x256) = Wtile(32x512) @ P =================
  const int dbase = wave * 32;
  v8f acc2[2][2];
#pragma unroll
  for (int mt = 0; mt < 2; ++mt)
#pragma unroll
    for (int dt = 0; dt < 2; ++dt)
#pragma unroll
      for (int i = 0; i < 8; ++i) acc2[mt][dt][i] = 0.f;

  const __bf16* pt_b = Pt + (size_t)b * D_ * N_;
#pragma unroll 1
  for (int kc = 0; kc < 16; ++kc) {
    const int kb = kc * 32;
    v16bf a[2];
    a[0] = afrag(&Wsh[ln][0], kb, hi);
    a[1] = afrag(&Wsh[16 + ln][0], kb, hi);
#pragma unroll
    for (int dt = 0; dt < 2; ++dt) {
      const int d = dbase + dt * 16 + ln;
      // B: lane=D col, element e -> K(=n) = kb + hi*16 + e (contiguous in Pt)
      v16bf bfv = *(const v16bf*)(pt_b + (size_t)d * N_ + kb + hi * 16);
#pragma unroll
      for (int mt = 0; mt < 2; ++mt)
        acc2[mt][dt] = __builtin_amdgcn_wmma_f32_16x16x32_bf16(
            false, a[mt], false, bfv, (short)0, acc2[mt][dt], false, false);
    }
  }

  // ---- store f32 embedding ----
#pragma unroll
  for (int mt = 0; mt < 2; ++mt)
#pragma unroll
    for (int r = 0; r < 8; ++r) {
      size_t rowoff = ((size_t)b * L_ + lt + mt * 16 + hi * 8 + r) * D_;
#pragma unroll
      for (int dt = 0; dt < 2; ++dt)
        Eout[rowoff + dbase + dt * 16 + ln] = acc2[mt][dt][r];
    }
}

// ---------------------------------------------------------------------------
extern "C" void kernel_launch(void* const* d_in, const int* in_sizes, int n_in,
                              void* d_out, int out_size, void* d_ws, size_t ws_size,
                              hipStream_t stream) {
  (void)in_sizes; (void)n_in; (void)out_size; (void)ws_size;
  const float* X    = (const float*)d_in[0];   // (B,L,D) f32
  const float* P    = (const float*)d_in[1];   // (B,N,D) f32
  const int*   lens = (const int*)d_in[2];     // (B,) i32

  float* out = (float*)d_out;
  float* emb = out;                              // (B,L,D)
  float* wts = out + (size_t)B_ * L_ * D_;       // (B,L,N)

  // workspace layout: [pnorm 32KB][Prow bf16 4MB][Pt bf16 4MB]
  char* ws = (char*)d_ws;
  float*  pnorm = (float*)ws;
  __bf16* Prow  = (__bf16*)(ws + 32 * 1024);
  __bf16* Pt    = (__bf16*)(ws + 32 * 1024 + (size_t)B_ * N_ * D_ * 2);

  prep_kernel<<<(B_ * N_) / 32, 256, 0, stream>>>(P, pnorm, Prow, Pt);
  fused_kernel<<<B_ * (L_ / 32), 256, 0, stream>>>(X, Prow, Pt, lens, pnorm,
                                                   wts, emb);
}
